// LinearWeightQuant_88570815578350
// MI455X (gfx1250) — matrix operand order repaired and checked by reading, not compiled, not run on hardware
//
#include <hip/hip_runtime.h>

typedef __attribute__((ext_vector_type(8)))  _Float16 v8h;
typedef __attribute__((ext_vector_type(16))) _Float16 v16h;
typedef __attribute__((ext_vector_type(8)))  float    v8f;

#define M_DIM 4096
#define K_DIM 4096
#define N_DIM 11008

constexpr int BM = 128;      // block tile M
constexpr int BN = 128;      // block tile N
constexpr int BK = 32;       // K step (one f16 WMMA depth)
constexpr int LDSTR = 40;    // LDS row stride in halves: 32 data + 8 pad (80B, 16B aligned)
constexpr int BUFH = BM * LDSTR;   // halves per buffer
constexpr int GRP = 8;       // N-block supertile width for L2 reuse

// y[m,n] = scale[n] * ( sum_k x[m,k]*w[n,k]  +  off[n]*rowsum[m] ) + bias[n]
__global__ __launch_bounds__(256)
void wq_linear_kernel(const float* __restrict__ x,
                      const int*   __restrict__ wq,
                      const float* __restrict__ scale,
                      const float* __restrict__ offs,
                      const float* __restrict__ bias,
                      float* __restrict__ out)
{
    // Double-buffered staging tiles (ping-pong): 2 * 10KB each.
    __shared__ __align__(16) _Float16 As[2 * BUFH];
    __shared__ __align__(16) _Float16 Bs[2 * BUFH];
    __shared__ float RsPart[256];
    __shared__ float RowSum[BM];

    constexpr int mBlocks = M_DIM / BM;   // 32
    constexpr int nBlocks = N_DIM / BN;   // 86

    // Supertiled block remap: walk all M blocks within a group of GRP N-blocks
    // so the weight stripe (GRP*128 cols * 4096 * 4B = 16MB) and the 2MB x
    // stripe stay resident in the 192MB L2.
    int bid = blockIdx.x;
    int grp = bid / (GRP * mBlocks);
    int rem = bid - grp * (GRP * mBlocks);
    int gw  = nBlocks - grp * GRP; if (gw > GRP) gw = GRP;
    int nBlk = grp * GRP + rem % gw;
    int mBlk = rem / gw;

    const int m0 = mBlk * BM;
    const int n0 = nBlk * BN;

    const int tid    = threadIdx.x;
    const int lane   = tid & 31;
    const int wid    = tid >> 5;       // 8 waves
    const int waveM  = wid >> 2;       // 0..1 : 64 M-rows each
    const int waveN  = wid & 3;        // 0..3 : 32 N-cols each
    const int laneLo = lane & 15;
    const int laneHi = lane >> 4;

    // Staging: two threads per row, each owns a 16-element half of the K slice.
    const int sRow  = tid >> 1;        // 0..127
    const int sHalf = tid & 1;

    const float* xRow = x  + (size_t)(m0 + sRow) * K_DIM + sHalf * 16;
    const int*   wRow = wq + (size_t)(n0 + sRow) * K_DIM + sHalf * 16;

    v8f c[4][2];
#pragma unroll
    for (int i = 0; i < 4; ++i)
#pragma unroll
        for (int j = 0; j < 2; ++j)
            c[i][j] = (v8f){0.f,0.f,0.f,0.f,0.f,0.f,0.f,0.f};

    float rsum = 0.f;   // partial row-sum of x over this thread's halves, all K

    // ---- software pipeline: preload K-slice 0 into registers ----
    float4 f0, f1, f2, f3;
    int4   w0, w1, w2, w3;
    {
        const float4* xg = (const float4*)xRow;
        f0 = xg[0]; f1 = xg[1]; f2 = xg[2]; f3 = xg[3];
        const int4* wg = (const int4*)wRow;
        w0 = wg[0]; w1 = wg[1]; w2 = wg[2]; w3 = wg[3];
    }

    for (int kt = 0; kt < K_DIM; kt += BK) {
        const int buf = (kt >> 5) & 1;          // ping-pong buffer
        _Float16* as = As + buf * BUFH;
        _Float16* bs = Bs + buf * BUFH;

        // ---- convert held registers -> f16, store to LDS buffer `buf` ----
        v8h h0 = { (_Float16)f0.x,(_Float16)f0.y,(_Float16)f0.z,(_Float16)f0.w,
                   (_Float16)f1.x,(_Float16)f1.y,(_Float16)f1.z,(_Float16)f1.w };
        v8h h1 = { (_Float16)f2.x,(_Float16)f2.y,(_Float16)f2.z,(_Float16)f2.w,
                   (_Float16)f3.x,(_Float16)f3.y,(_Float16)f3.z,(_Float16)f3.w };
        *(v8h*)&as[sRow * LDSTR + sHalf * 16 + 0] = h0;
        *(v8h*)&as[sRow * LDSTR + sHalf * 16 + 8] = h1;
        rsum += f0.x + f0.y + f0.z + f0.w + f1.x + f1.y + f1.z + f1.w
              + f2.x + f2.y + f2.z + f2.w + f3.x + f3.y + f3.z + f3.w;

        v8h g0 = { (_Float16)w0.x,(_Float16)w0.y,(_Float16)w0.z,(_Float16)w0.w,
                   (_Float16)w1.x,(_Float16)w1.y,(_Float16)w1.z,(_Float16)w1.w };
        v8h g1 = { (_Float16)w2.x,(_Float16)w2.y,(_Float16)w2.z,(_Float16)w2.w,
                   (_Float16)w3.x,(_Float16)w3.y,(_Float16)w3.z,(_Float16)w3.w };
        *(v8h*)&bs[sRow * LDSTR + sHalf * 16 + 0] = g0;
        *(v8h*)&bs[sRow * LDSTR + sHalf * 16 + 8] = g1;

        // Single barrier per K-step: publishes this buffer's stores AND, being
        // program-order after every wave's reads of the *other* buffer last
        // iteration, licenses its reuse next iteration.
        __syncthreads();

        // ---- issue next K-slice's global loads; they and the converts above
        //      overlap the ds_loads + WMMAs below (uniform branch: EXEC stays
        //      all-ones for WMMA) ----
        if (kt + BK < K_DIM) {
            const float4* xg = (const float4*)(xRow + kt + BK);
            f0 = xg[0]; f1 = xg[1]; f2 = xg[2]; f3 = xg[3];
            const int4* wg = (const int4*)(wRow + kt + BK);
            w0 = wg[0]; w1 = wg[1]; w2 = wg[2]; w3 = wg[3];
            if (kt + 2 * BK < K_DIM) {
                __builtin_prefetch((const void*)(xRow + kt + 2 * BK), 0, 3);
                __builtin_prefetch((const void*)(wRow + kt + 2 * BK), 0, 3);
            }
        }

        // ---- load WMMA fragments (ISA 7.12.2 wave32 16-bit layouts) ----
        // A 16x32: lane<16 holds K {0..7,16..23}; lane>=16 holds {8..15,24..31}
        v16h a[4];
#pragma unroll
        for (int i = 0; i < 4; ++i) {
            int base = (waveM * 64 + i * 16 + laneLo) * LDSTR + laneHi * 8;
            v8h lo = *(const v8h*)&as[base];
            v8h hi = *(const v8h*)&as[base + 16];
            a[i] = __builtin_shufflevector(lo, hi,
                     0,1,2,3,4,5,6,7,8,9,10,11,12,13,14,15);
        }
        // B 32x16: lane<16 holds K {0..7,16..23}; lane>=16 holds {8..15,24..31}
        v16h b[2];
#pragma unroll
        for (int j = 0; j < 2; ++j) {
            int base = (waveN * 32 + j * 16 + laneLo) * LDSTR + laneHi * 8;
            v8h lo = *(const v8h*)&bs[base];
            v8h hi = *(const v8h*)&bs[base + 16];
            b[j] = __builtin_shufflevector(lo, hi,
                     0,1,2,3,4,5,6,7,8,9,10,11,12,13,14,15);
        }

        // ---- 8 WMMAs per K step ----
#pragma unroll
        for (int i = 0; i < 4; ++i)
#pragma unroll
            for (int j = 0; j < 2; ++j)
                c[i][j] = __builtin_amdgcn_wmma_f32_16x16x32_f16(
                    false, a[i], false, b[j], (short)0, c[i][j], false, false);
    }

    // ---- finalize rowsum[m] for this block's 128 rows ----
    RsPart[tid] = rsum;
    __syncthreads();
    if (tid < BM)
        RowSum[tid] = RsPart[2 * tid] + RsPart[2 * tid + 1];
    __syncthreads();

    // ---- epilogue: y = scale*(G + off*rowsum) + bias ----
    // C/D layout: lanes 0-15 -> M = r, lanes 16-31 -> M = r+8; N = lane&15.
#pragma unroll
    for (int j = 0; j < 2; ++j) {
        int n = n0 + waveN * 32 + j * 16 + laneLo;
        float sc = scale[n];
        float of = offs[n];
        float bi = bias[n];
#pragma unroll
        for (int i = 0; i < 4; ++i) {
#pragma unroll
            for (int r = 0; r < 8; ++r) {
                int ml = waveM * 64 + i * 16 + laneHi * 8 + r;
                float v = sc * (c[i][j][r] + of * RowSum[ml]) + bi;
                out[(size_t)(m0 + ml) * N_DIM + n] = v;
            }
        }
    }
}

extern "C" void kernel_launch(void* const* d_in, const int* in_sizes, int n_in,
                              void* d_out, int out_size, void* d_ws, size_t ws_size,
                              hipStream_t stream) {
    (void)in_sizes; (void)n_in; (void)out_size; (void)d_ws; (void)ws_size;
    const float* x     = (const float*)d_in[0];
    const int*   wq    = (const int*)d_in[1];
    const float* scale = (const float*)d_in[2];
    const float* offs  = (const float*)d_in[3];
    const float* bias  = (const float*)d_in[4];
    float* out = (float*)d_out;

    constexpr int mBlocks = M_DIM / BM;   // 32
    constexpr int nBlocks = N_DIM / BN;   // 86
    dim3 grid(mBlocks * nBlocks);         // 2752 blocks
    dim3 block(256);                      // 8 waves (wave32)
    wq_linear_kernel<<<grid, block, 0, stream>>>(x, wq, scale, offs, bias, out);
}
